// GMTRouterModel_35390530519326
// MI455X (gfx1250) — matrix operand order repaired and checked by reading, not compiled, not run on hardware
//
#include <hip/hip_runtime.h>
#include <hip/hip_bf16.h>

// ---------------------------------------------------------------------------
// GMT router model on MI455X (gfx1250).
// Dense matmuls: v_wmma_f32_16x16x32_bf16, wave32, one wave per 16-row block
// x NT*16 columns (NT accumulators, A fragment reused across NT B tiles).
// NT is a template parameter (8 or 4) so the K-loop body is branch-free.
// Edge softmax-aggregation uses ordered-uint atomicMax + f32 atomicAdd.
// Workspace requirement: ~625 MB.
// ---------------------------------------------------------------------------

typedef __attribute__((ext_vector_type(16))) __bf16 v16bf;
typedef __attribute__((ext_vector_type(8)))  float  v8f;

#define HID  128
#define NU_  50000
#define NQ_  100000
#define NL_  64
#define NTOT_ (NU_ + NQ_ + NL_)
#define BSZ  8192

__device__ __forceinline__ unsigned encf(float f) {
  unsigned u = __float_as_uint(f);
  return (u >> 31) ? ~u : (u | 0x80000000u);
}
__device__ __forceinline__ float decf(unsigned u) {
  u = (u >> 31) ? (u & 0x7FFFFFFFu) : ~u;
  return __uint_as_float(u);
}

// ---------------- weight prep ----------------
// W[K x N] f32 (row major) -> Wt[N x K] bf16
__global__ void k_transpose_bf16(const float* __restrict__ W, __bf16* __restrict__ Wt,
                                 int K, int N) {
  int idx = blockIdx.x * 256 + threadIdx.x;
  if (idx >= K * N) return;
  int n = idx / K, k = idx - n * K;
  Wt[idx] = (__bf16)W[(size_t)k * N + n];
}

// R[4][32][32] (h,d,e) -> block-diagonal Wt[128][128] bf16 (n-major, [n][k])
// out[n = h*32+e] from in[k = h*32+d]:  Wt[n][k] = R[h][d][e]
__global__ void k_blockdiag_bf16(const float* __restrict__ R, __bf16* __restrict__ Wt) {
  int idx = blockIdx.x * 256 + threadIdx.x;  // 16384 threads
  if (idx >= 128 * 128) return;
  int n = idx >> 7, k = idx & 127;
  int hn = n >> 5, hk = k >> 5;
  float v = (hn == hk) ? R[hn * 1024 + (k & 31) * 32 + (n & 31)] : 0.0f;
  Wt[idx] = (__bf16)v;
}

// ---------------- WMMA GEMM ----------------
// C[M x N] = A[M x K](f32, lda) * Wt[N x K](bf16) + bias ; act: 0 none, 1 relu
// grid = (ceil(M/128), ceil(N/(NT*16))), block = 256.
// Each wave: one 16-row block x NT 16x16 WMMA tiles (branch-free K-loop body).
// M is always a multiple of 16 and N a multiple of NT*16 at every call site.
template <int NT>
__global__ __launch_bounds__(256) void k_gemm(
    const float* __restrict__ A, int lda,
    const __bf16* __restrict__ Wt,
    const float* __restrict__ bias,
    float* __restrict__ C, int ldc,
    int M, int K, int act) {
  int lane = threadIdx.x & 31;
  int wave = threadIdx.x >> 5;
  int m0 = (blockIdx.x * 8 + wave) * 16;
  if (m0 >= M) return;
  int nBase = blockIdx.y * (NT * 16);
  int half = lane >> 4;
  int l15  = lane & 15;
  const float*  arow = A + (size_t)(m0 + l15) * lda;
  const __bf16* bcol = Wt + (size_t)(nBase + l15) * K + 16 * half;

  v8f acc[NT];
#pragma unroll
  for (int t = 0; t < NT; ++t) acc[t] = (v8f){};

  for (int k0 = 0; k0 < K; k0 += 32) {
    // A 16-bit fragment: a[s] = arow[k0 + 16*half + s]
    // => two contiguous 8-float runs at k0+16*half and k0+16*half+8.
    v16bf a;
    {
      const float4* p0 = (const float4*)(arow + k0 + 16 * half);
      const float4* p1 = (const float4*)(arow + k0 + 16 * half + 8);
      float4 f0 = p0[0], f1 = p0[1], f2 = p1[0], f3 = p1[1];
      a[0]  = (__bf16)f0.x; a[1]  = (__bf16)f0.y; a[2]  = (__bf16)f0.z; a[3]  = (__bf16)f0.w;
      a[4]  = (__bf16)f1.x; a[5]  = (__bf16)f1.y; a[6]  = (__bf16)f1.z; a[7]  = (__bf16)f1.w;
      a[8]  = (__bf16)f2.x; a[9]  = (__bf16)f2.y; a[10] = (__bf16)f2.z; a[11] = (__bf16)f2.w;
      a[12] = (__bf16)f3.x; a[13] = (__bf16)f3.y; a[14] = (__bf16)f3.z; a[15] = (__bf16)f3.w;
    }
#pragma unroll
    for (int t = 0; t < NT; ++t) {
      // B 16-bit fragment: b[2j(+1)] = Wt[n0+l15][k0 + 16*half + 2j(+1)]
      // => 16 contiguous bf16 (32 B, 32B-aligned) -> vector load.
      v16bf b = *(const v16bf*)(bcol + (size_t)t * 16 * K + k0);
      acc[t] = __builtin_amdgcn_wmma_f32_16x16x32_bf16(false, a, false, b,
                                                       (short)0, acc[t], false, false);
    }
  }

#pragma unroll
  for (int t = 0; t < NT; ++t) {
    int n = nBase + t * 16 + l15;
    float bv = bias ? bias[n] : 0.0f;
#pragma unroll
    for (int r = 0; r < 8; ++r) {
      int m = m0 + r + 8 * half;
      float v = acc[t][r] + bv;
      if (act == 1) v = fmaxf(v, 0.0f);
      C[(size_t)m * ldc + n] = v;
    }
  }
}

// ---------------- elementwise ----------------
__global__ void k_fill_u32(unsigned* __restrict__ p, unsigned v, int n) {
  int i = blockIdx.x * 256 + threadIdx.x;
  if (i < n) p[i] = v;
}

__global__ void k_gelu(float* __restrict__ x, size_t n) {
  size_t i = (size_t)blockIdx.x * 256 + threadIdx.x;
  if (i >= n) return;
  float v = x[i];
  x[i] = 0.5f * v * (1.0f + erff(v * 0.70710678118654752f));
}

// ---------------- edge phases ----------------
__global__ void k_edge_logits(const float* __restrict__ kqv,
                              const float* __restrict__ krel,
                              const int* __restrict__ esrc,
                              const int* __restrict__ edst,
                              const float* __restrict__ prel,
                              float* __restrict__ logits,  // [E][4]
                              unsigned* __restrict__ segmax,
                              int E, int dstOff) {
  int idx = blockIdx.x * 256 + threadIdx.x;
  if (idx >= E * 4) return;
  int e = idx >> 2, h = idx & 3;
  int s = esrc[e], d = edst[e];
  const float* qp = kqv + (size_t)(dstOff + d) * 384 + 128 + h * 32;
  const float* kp = krel + (size_t)s * 128 + h * 32;
  float acc = 0.0f;
#pragma unroll
  for (int i = 0; i < 32; ++i) acc += qp[i] * kp[i];
  float a = acc * prel[h] * 0.17677669529663687f;  // / sqrt(32)
  logits[idx] = a;
  atomicMax(&segmax[(size_t)(dstOff + d) * 4 + h], encf(a));
}

__global__ void k_edge_exp(float* __restrict__ logits,
                           const int* __restrict__ edst,
                           const unsigned* __restrict__ segmax,
                           float* __restrict__ segsum,
                           int E, int dstOff) {
  int idx = blockIdx.x * 256 + threadIdx.x;
  if (idx >= E * 4) return;
  int e = idx >> 2, h = idx & 3;
  size_t seg = (size_t)(dstOff + edst[e]) * 4 + h;
  float m = decf(segmax[seg]);
  if (!isfinite(m)) m = 0.0f;
  float ex = expf(logits[idx] - m);
  logits[idx] = ex;
  atomicAdd(&segsum[seg], ex);
}

__global__ void k_edge_agg(const float* __restrict__ logits,
                           const int* __restrict__ esrc,
                           const int* __restrict__ edst,
                           const float* __restrict__ vrel,
                           const float* __restrict__ segsum,
                           float* __restrict__ agg,
                           int E, int dstOff) {
  size_t idx = (size_t)blockIdx.x * 256 + threadIdx.x;
  if (idx >= (size_t)E * 128) return;
  int e = (int)(idx >> 7);
  int d = (int)(idx & 127);
  int h = d >> 5;
  int seg = dstOff + edst[e];
  float w = logits[(size_t)e * 4 + h] / (segsum[(size_t)seg * 4 + h] + 1e-16f);
  atomicAdd(&agg[(size_t)seg * 128 + d], w * vrel[(size_t)esrc[e] * 128 + d]);
}

// ---------------- node update: skip-gate + layernorm (wave per row) ----------
__global__ __launch_bounds__(256) void k_update_ln(
    const float* __restrict__ o, float* __restrict__ x,
    const float* __restrict__ skipp,
    const float* __restrict__ gamma, const float* __restrict__ beta, int n) {
  int wave = threadIdx.x >> 5, lane = threadIdx.x & 31;
  int row = blockIdx.x * 8 + wave;
  if (row >= n) return;
  float g = 1.0f / (1.0f + expf(-skipp[0]));
  float v[4], s = 0.0f;
#pragma unroll
  for (int j = 0; j < 4; ++j) {
    int d = lane + 32 * j;
    size_t idx = (size_t)row * 128 + d;
    v[j] = g * o[idx] + (1.0f - g) * x[idx];
    s += v[j];
  }
#pragma unroll
  for (int off = 16; off; off >>= 1) s += __shfl_xor(s, off, 32);
  float mean = s * (1.0f / 128.0f);
  float sv = 0.0f;
#pragma unroll
  for (int j = 0; j < 4; ++j) { float t = v[j] - mean; sv += t * t; }
#pragma unroll
  for (int off = 16; off; off >>= 1) sv += __shfl_xor(sv, off, 32);
  float inv = rsqrtf(sv * (1.0f / 128.0f) + 1e-5f);
#pragma unroll
  for (int j = 0; j < 4; ++j) {
    int d = lane + 32 * j;
    x[(size_t)row * 128 + d] = (v[j] - mean) * inv * gamma[d] + beta[d];
  }
}

// ---------------- prediction head ----------------
__global__ void k_gather(const float* __restrict__ x_all, const int* __restrict__ ids,
                         int idOff, float* __restrict__ out, int B) {
  int idx = blockIdx.x * 256 + threadIdx.x;
  if (idx >= B * 128) return;
  int i = idx >> 7, d = idx & 127;
  out[idx] = x_all[(size_t)(idOff + ids[i]) * 128 + d];
}

// wave per (b, h): 2-key attention over {user, llm} contexts
__global__ __launch_bounds__(256) void k_attn(
    const float* __restrict__ qh, const float* __restrict__ kU,
    const float* __restrict__ kL, const float* __restrict__ vU,
    const float* __restrict__ vL, float* __restrict__ o, int B) {
  int wave = threadIdx.x >> 5, lane = threadIdx.x & 31;
  int idx = blockIdx.x * 8 + wave;
  if (idx >= B * 4) return;
  int b = idx >> 2, h = idx & 3;
  size_t base = (size_t)b * 128 + h * 32 + lane;
  float q = qh[base];
  float s0 = q * kU[base];
  float s1 = q * kL[base];
#pragma unroll
  for (int off = 16; off; off >>= 1) {
    s0 += __shfl_xor(s0, off, 32);
    s1 += __shfl_xor(s1, off, 32);
  }
  s0 *= 0.17677669529663687f;
  s1 *= 0.17677669529663687f;
  float mx = fmaxf(s0, s1);
  float e0 = expf(s0 - mx), e1 = expf(s1 - mx);
  float inv = 1.0f / (e0 + e1);
  o[base] = (e0 * vU[base] + e1 * vL[base]) * inv;
}

__global__ void k_final(const float* __restrict__ h1, const float* __restrict__ W2,
                        const float* __restrict__ b2, float* __restrict__ out, int B) {
  int b = blockIdx.x * 256 + threadIdx.x;
  if (b >= B) return;
  float acc = b2[0];
#pragma unroll
  for (int i = 0; i < 64; ++i) acc += h1[(size_t)b * 64 + i] * W2[i];
  out[b] = acc;
}

// ---------------------------------------------------------------------------
extern "C" void kernel_launch(void* const* d_in, const int* in_sizes, int n_in,
                              void* d_out, int out_size, void* d_ws, size_t ws_size,
                              hipStream_t stream) {
  (void)in_sizes; (void)n_in; (void)out_size; (void)ws_size;
  const int NU = NU_, NQ = NQ_, NL = NL_, NTOT = NTOT_, B = BSZ;

  auto F = [&](int i) { return (const float*)d_in[i]; };
  auto I = [&](int i) { return (const int*)d_in[i]; };

  // ---- input leaf indices (JAX pytree: dict keys sorted) ----
  // 0..2: x_user, x_query, x_llm
  // gnn_params: 'layers' then 'proj'.
  //   layer leaves (base lb): a_rel[lq,ql,qu,uq]=lb+0..3,
  //     kqv (llm,query,user; W,b pairs)=lb+4..9, ln=lb+10..15,
  //     m_rel=lb+16..19, out=lb+20..25, p_rel=lb+26..29, skip=lb+30..32
  //   layer0 lb=3, layer1 lb=36; proj leaves 69..74
  // pred: in_proj W/b=75/76, out_proj=77/78, w1=79/80, w2=81/82
  // edges uq/qu/ql/lq = 83..86; user/query/llm ids = 87..89

  // processing order: types (user, query, llm), rels (uq, qu, ql, lq)
  int tOff[3]  = {0, NU, NU + NQ};
  int tCnt[3]  = {NU, NQ, NL};
  int tSort[3] = {2, 1, 0};  // position in sorted (llm,query,user)

  int rSrcT[4]    = {0, 1, 1, 2};
  int rDstT[4]    = {1, 0, 2, 1};
  int rE[4]       = {100000, 100000, 200000, 200000};
  int rEdgeIdx[4] = {83, 84, 85, 86};
  int rSort[4]    = {3, 2, 1, 0};  // position in sorted (lq,ql,qu,uq)
  int rBase[4]    = {0, 100000, 200000, 400000};

  // ---- workspace arena ----
  char* wsb = (char*)d_ws;
  size_t off = 0;
  auto alloc = [&](size_t bytes) -> void* {
    off = (off + 255) & ~(size_t)255;
    void* p = wsb + off;
    off += bytes;
    return p;
  };
  float*    x_all  = (float*)alloc((size_t)NTOT * 128 * 4);
  float*    kqv    = (float*)alloc((size_t)NTOT * 384 * 4);
  float*    krel   = (float*)alloc((size_t)100000 * 128 * 4);
  float*    vrel[4];
  for (int r = 0; r < 4; ++r) vrel[r] = (float*)alloc((size_t)tCnt[rSrcT[r]] * 128 * 4);
  float*    logits = (float*)alloc((size_t)600000 * 4 * 4);
  unsigned* segmax = (unsigned*)alloc((size_t)NTOT * 4 * 4);
  float*    segsum = (float*)alloc((size_t)NTOT * 4 * 4);
  float*    agg    = (float*)alloc((size_t)NTOT * 128 * 4);

  __bf16* proj_t[3];
  __bf16* kqv_t[2][3];
  __bf16* arel_t[2][4];
  __bf16* mrel_t[2][4];
  __bf16* out_t[2][3];
  for (int t = 0; t < 3; ++t) proj_t[t] = (__bf16*)alloc(128 * 128 * 2);
  for (int L = 0; L < 2; ++L) {
    for (int t = 0; t < 3; ++t) kqv_t[L][t] = (__bf16*)alloc(384 * 128 * 2);
    for (int t = 0; t < 3; ++t) out_t[L][t] = (__bf16*)alloc(128 * 128 * 2);
    for (int r = 0; r < 4; ++r) arel_t[L][r] = (__bf16*)alloc(128 * 128 * 2);
    for (int r = 0; r < 4; ++r) mrel_t[L][r] = (__bf16*)alloc(128 * 128 * 2);
  }
  __bf16* inproj_t  = (__bf16*)alloc(384 * 128 * 2);
  __bf16* outproj_t = (__bf16*)alloc(128 * 128 * 2);
  __bf16* w1_t      = (__bf16*)alloc(64 * 128 * 2);

  float* Qin = (float*)alloc((size_t)B * 128 * 4);
  float* Uin = (float*)alloc((size_t)B * 128 * 4);
  float* Lin = (float*)alloc((size_t)B * 128 * 4);
  float* qh  = (float*)alloc((size_t)B * 128 * 4);
  float* kU  = (float*)alloc((size_t)B * 128 * 4);
  float* kL  = (float*)alloc((size_t)B * 128 * 4);
  float* vU  = (float*)alloc((size_t)B * 128 * 4);
  float* vL  = (float*)alloc((size_t)B * 128 * 4);
  float* oa  = (float*)alloc((size_t)B * 128 * 4);
  float* o2  = (float*)alloc((size_t)B * 128 * 4);
  float* h1  = (float*)alloc((size_t)B * 64 * 4);

  float* obuf = kqv;  // out-proj scratch aliases the dead kqv arena

  // N must be a multiple of 128 (k_gemm<8>) or exactly 64 (k_gemm<4>).
  auto gemm = [&](const float* A, int lda, const __bf16* Wt, const float* bias,
                  float* C, int ldc, int M, int N, int K, int act) {
    if (N % 128 == 0) {
      dim3 g((M + 127) / 128, N / 128);
      k_gemm<8><<<g, 256, 0, stream>>>(A, lda, Wt, bias, C, ldc, M, K, act);
    } else {
      dim3 g((M + 127) / 128, (N + 63) / 64);
      k_gemm<4><<<g, 256, 0, stream>>>(A, lda, Wt, bias, C, ldc, M, K, act);
    }
  };
  auto tr = [&](const float* W, __bf16* Wt, int K, int N) {
    int n = K * N;
    k_transpose_bf16<<<(n + 255) / 256, 256, 0, stream>>>(W, Wt, K, N);
  };

  // ---- weight prep (bf16, transposed) ----
  for (int t = 0; t < 3; ++t) tr(F(69 + 2 * tSort[t]), proj_t[t], 128, 128);
  for (int L = 0; L < 2; ++L) {
    int lb = 3 + 33 * L;
    for (int t = 0; t < 3; ++t) {
      tr(F(lb + 4 + 2 * tSort[t]), kqv_t[L][t], 128, 384);
      tr(F(lb + 20 + 2 * tSort[t]), out_t[L][t], 128, 128);
    }
    for (int r = 0; r < 4; ++r) {
      k_blockdiag_bf16<<<64, 256, 0, stream>>>(F(lb + 0 + rSort[r]), arel_t[L][r]);
      k_blockdiag_bf16<<<64, 256, 0, stream>>>(F(lb + 16 + rSort[r]), mrel_t[L][r]);
    }
  }
  tr(F(75), inproj_t, 128, 384);
  tr(F(77), outproj_t, 128, 128);
  tr(F(79), w1_t, 128, 64);

  // ---- initial per-type projection into x_all ----
  for (int t = 0; t < 3; ++t)
    gemm(F(t), 128, proj_t[t], F(69 + 2 * tSort[t] + 1),
         x_all + (size_t)tOff[t] * 128, 128, tCnt[t], 128, 128, 0);

  // ---- HGT layers ----
  for (int L = 0; L < 2; ++L) {
    int lb = 3 + 33 * L;
    // kqv projection (N=384)
    for (int t = 0; t < 3; ++t)
      gemm(x_all + (size_t)tOff[t] * 128, 128, kqv_t[L][t], F(lb + 4 + 2 * tSort[t] + 1),
           kqv + (size_t)tOff[t] * 384, 384, tCnt[t], 384, 128, 0);

    hipMemsetAsync(segsum, 0, (size_t)NTOT * 4 * 4, stream);
    hipMemsetAsync(agg, 0, (size_t)NTOT * 128 * 4, stream);
    k_fill_u32<<<(NTOT * 4 + 255) / 256, 256, 0, stream>>>(segmax, 0x007FFFFFu, NTOT * 4);

    // phase 1: rel transforms + logits + segment max
    for (int r = 0; r < 4; ++r) {
      int sT = rSrcT[r], dT = rDstT[r];
      const float* kA = kqv + (size_t)tOff[sT] * 384;        // k slice
      const float* vA = kqv + (size_t)tOff[sT] * 384 + 256;  // v slice
      gemm(kA, 384, arel_t[L][r], nullptr, krel, 128, tCnt[sT], 128, 128, 0);
      gemm(vA, 384, mrel_t[L][r], nullptr, vrel[r], 128, tCnt[sT], 128, 128, 0);
      const int* esrc = I(rEdgeIdx[r]);
      const int* edst = esrc + rE[r];
      int n = rE[r] * 4;
      k_edge_logits<<<(n + 255) / 256, 256, 0, stream>>>(
          kqv, krel, esrc, edst, F(lb + 26 + rSort[r]),
          logits + (size_t)rBase[r] * 4, segmax, rE[r], tOff[dT]);
    }
    // phase 2: exp + segment sum
    for (int r = 0; r < 4; ++r) {
      const int* esrc = I(rEdgeIdx[r]);
      const int* edst = esrc + rE[r];
      int n = rE[r] * 4;
      k_edge_exp<<<(n + 255) / 256, 256, 0, stream>>>(
          logits + (size_t)rBase[r] * 4, edst, segmax, segsum, rE[r], tOff[rDstT[r]]);
    }
    // phase 3: normalized message aggregation
    for (int r = 0; r < 4; ++r) {
      const int* esrc = I(rEdgeIdx[r]);
      const int* edst = esrc + rE[r];
      size_t n = (size_t)rE[r] * 128;
      k_edge_agg<<<(unsigned)((n + 255) / 256), 256, 0, stream>>>(
          logits + (size_t)rBase[r] * 4, esrc, edst, vrel[r], segsum, agg,
          rE[r], tOff[rDstT[r]]);
    }
    // gelu(agg) -> out-proj -> skip-gate + layernorm (writes x_all in place)
    size_t ntEl = (size_t)NTOT * 128;
    k_gelu<<<(unsigned)((ntEl + 255) / 256), 256, 0, stream>>>(agg, ntEl);
    for (int t = 0; t < 3; ++t)
      gemm(agg + (size_t)tOff[t] * 128, 128, out_t[L][t], F(lb + 20 + 2 * tSort[t] + 1),
           obuf + (size_t)tOff[t] * 128, 128, tCnt[t], 128, 128, 0);
    for (int t = 0; t < 3; ++t)
      k_update_ln<<<(tCnt[t] + 7) / 8, 256, 0, stream>>>(
          obuf + (size_t)tOff[t] * 128, x_all + (size_t)tOff[t] * 128,
          F(lb + 30 + tSort[t]), F(lb + 10 + 2 * tSort[t]),
          F(lb + 10 + 2 * tSort[t] + 1), tCnt[t]);
  }

  // ---- prediction head ----
  int nB = B * 128;
  k_gather<<<(nB + 255) / 256, 256, 0, stream>>>(x_all, I(88), NU, Qin, B);       // query
  k_gather<<<(nB + 255) / 256, 256, 0, stream>>>(x_all, I(87), 0, Uin, B);        // user
  k_gather<<<(nB + 255) / 256, 256, 0, stream>>>(x_all, I(89), NU + NQ, Lin, B);  // llm

  const float* bi = F(76);
  gemm(Qin, 128, inproj_t,             bi,       qh, 128, B, 128, 128, 0);
  gemm(Uin, 128, inproj_t + 128 * 128, bi + 128, kU, 128, B, 128, 128, 0);
  gemm(Lin, 128, inproj_t + 128 * 128, bi + 128, kL, 128, B, 128, 128, 0);
  gemm(Uin, 128, inproj_t + 256 * 128, bi + 256, vU, 128, B, 128, 128, 0);
  gemm(Lin, 128, inproj_t + 256 * 128, bi + 256, vL, 128, B, 128, 128, 0);

  k_attn<<<(B * 4 + 7) / 8, 256, 0, stream>>>(qh, kU, kL, vU, vL, oa, B);

  gemm(oa, 128, outproj_t, F(78), o2, 128, B, 128, 128, 0);
  gemm(o2, 128, w1_t, F(80), h1, 64, B, 64, 128, 1);  // relu
  k_final<<<(B + 255) / 256, 256, 0, stream>>>(h1, F(81), F(82), (float*)d_out, B);
}
